// GraphLayer_42356967473339
// MI455X (gfx1250) — compile-verified, matrix operand order checked by reading, not run on hardware
//
#include <hip/hip_runtime.h>
#include <hip/hip_bf16.h>

// ---------------- dimensions (fixed by the reference) ----------------
constexpr int BN   = 32;    // batch
constexpr int NN   = 512;   // sequence
constexpr int DM   = 512;   // d_model
constexpr int NH   = 8;     // heads
constexpr int DH   = 64;    // head dim
constexpr int DI   = 2048;  // inner MLP dim
constexpr int ROWS = BN * NN;  // 16384 flattened rows

typedef __attribute__((ext_vector_type(16))) __bf16 v16bf;
typedef __attribute__((ext_vector_type(8)))  __bf16 v8bf;
typedef __attribute__((ext_vector_type(8)))  float  v8f;

// ---------------- helpers ----------------
__device__ inline __bf16 f32_to_bf16(float f) {
  union { float f; unsigned u; } in; in.f = f;
  unsigned u = in.u;
  u += 0x7FFFu + ((u >> 16) & 1u);          // round-to-nearest-even
  union { unsigned short s; __bf16 b; } out;
  out.s = (unsigned short)(u >> 16);
  return out.b;
}

// Load a 16x32 bf16 WMMA operand (A layout; B uses the same pattern with
// rows == output columns).  p points at row 0 of the 16-row tile,
// ld = row stride in elements, k0 = starting K (multiple of 32, 16B aligned).
__device__ inline v16bf load_frag(const __bf16* p, int ld, int k0) {
  int lane = threadIdx.x & 31;
  int halfo = (lane >> 4) * 8;      // K sub-offset per lane half
  int idx  = lane & 15;             // row (A) / column (B)
  const __bf16* row = p + (size_t)idx * ld + k0 + halfo;
  union { v16bf v; v8bf h[2]; } u;
  u.h[0] = *(const v8bf*)(row);
  u.h[1] = *(const v8bf*)(row + 16);
  return u.v;
}

__device__ inline v8f wmma_bf16(v16bf a, v16bf b, v8f c) {
  return __builtin_amdgcn_wmma_f32_16x16x32_bf16(
      false, a, false, b, (short)0, c, false, false);
}

// ---------------- elementwise f32 -> bf16 ----------------
__global__ void cvt_bf16_kernel(const float* __restrict__ s, __bf16* __restrict__ d, int n) {
  int i = blockIdx.x * blockDim.x + threadIdx.x;
  if (i < n) d[i] = f32_to_bf16(s[i]);
}

// ---------------- LayerNorm over DM=512, emit bf16 ----------------
__global__ __launch_bounds__(256) void ln_kernel(const float* __restrict__ X,
                                                 const float* __restrict__ w,
                                                 const float* __restrict__ bb,
                                                 __bf16* __restrict__ Y) {
  int row = blockIdx.x;
  const float* x = X + (size_t)row * DM;
  int t = threadIdx.x;
  float a0 = x[t], a1 = x[t + 256];
  float s = a0 + a1, s2 = a0 * a0 + a1 * a1;
  for (int m = 1; m < 32; m <<= 1) { s += __shfl_xor(s, m); s2 += __shfl_xor(s2, m); }
  __shared__ float ssum[8], ssq[8];
  int wid = t >> 5;
  if ((t & 31) == 0) { ssum[wid] = s; ssq[wid] = s2; }
  __syncthreads();
  float tot = 0.f, tot2 = 0.f;
  for (int i = 0; i < 8; i++) { tot += ssum[i]; tot2 += ssq[i]; }
  float mu  = tot * (1.0f / DM);
  float var = tot2 * (1.0f / DM) - mu * mu;
  float rs  = rsqrtf(var + 1e-5f);
  __bf16* y = Y + (size_t)row * DM;
  y[t]       = f32_to_bf16((a0 - mu) * rs * w[t]       + bb[t]);
  y[t + 256] = f32_to_bf16((a1 - mu) * rs * w[t + 256] + bb[t + 256]);
}

// ---------------- QKV GEMM: [ROWS,512] x [1536,512]^T, scatter to Q/K/Vt ----------------
__global__ __launch_bounds__(32) void gemm_qkv_kernel(const __bf16* __restrict__ A,
                                                      const __bf16* __restrict__ W,
                                                      const float* __restrict__ bias,
                                                      __bf16* __restrict__ Qo,
                                                      __bf16* __restrict__ Ko,
                                                      __bf16* __restrict__ Vto) {
  int m0 = blockIdx.y * 16;
  int n0 = blockIdx.x * 64;
  v8f acc[4] = {};
  for (int k = 0; k < DM; k += 32) {
    v16bf a = load_frag(A + (size_t)m0 * DM, DM, k);
    for (int j = 0; j < 4; j++)
      acc[j] = wmma_bf16(a, load_frag(W + (size_t)(n0 + j * 16) * DM, DM, k), acc[j]);
  }
  int lane = threadIdx.x & 31, halfb = lane >> 4, idx = lane & 15;
  for (int j = 0; j < 4; j++)
    for (int v = 0; v < 8; v++) {
      int col = n0 + j * 16 + idx;
      int m = m0 + halfb * 8 + v;
      int b = m >> 9, n = m & (NN - 1);
      float val = acc[j][v] + bias[col];
      int h = col / 192, w = col - h * 192, ck = w >> 6, d = w & 63;
      __bf16 bv = f32_to_bf16(val);
      if (ck == 1)      Qo[((size_t)(b * NH + h) * NN + n) * DH + d] = bv;  // Z3 = Q
      else if (ck == 2) Ko[((size_t)(b * NH + h) * NN + n) * DH + d] = bv;  // Z4 = K
      else              Vto[((size_t)(b * NH + h) * DH + d) * NN + n] = bv; // Z2 = V (transposed)
    }
}

// ---------------- fused attention (flash-style, one wave = 16 query rows x 1 head) ----------------
__global__ __launch_bounds__(32) void attn_kernel(const __bf16* __restrict__ Q,
                                                  const __bf16* __restrict__ K,
                                                  const __bf16* __restrict__ Vt,
                                                  const float* __restrict__ Dd,
                                                  const float* __restrict__ NMd,
                                                  const float* __restrict__ MSK,
                                                  const float* __restrict__ gamma,
                                                  const int* __restrict__ ra_p,
                                                  const int* __restrict__ rb_p,
                                                  __bf16* __restrict__ attn_out) {
  int lane = threadIdx.x & 31, halfb = lane >> 4, idx = lane & 15;
  int r0 = blockIdx.x * 16;
  int h  = blockIdx.y, b = blockIdx.z;
  int ra = ra_p[0], rb = rb_p[0];
  float g[16];
  for (int i = 0; i < 16; i++) g[i] = gamma[i];

  const __bf16* Qh = Q  + ((size_t)(b * NH + h) * NN + r0) * DH;
  const __bf16* Kh = K  + (size_t)(b * NH + h) * NN * DH;
  const __bf16* Vh = Vt + (size_t)(b * NH + h) * DH * NN;

  v16bf qa0 = load_frag(Qh, DH, 0);
  v16bf qa1 = load_frag(Qh, DH, 32);

  float Mx[8], L[8];
  int regr[8];
  v8f acc[4] = {};
  for (int v = 0; v < 8; v++) {
    Mx[v] = -1e30f; L[v] = 0.f;
    int r = r0 + halfb * 8 + v;
    regr[v] = (r >= ra) + (r >= rb);
  }

  __shared__ __bf16 plds[16 * 32];
  const float scale = 0.125f;  // 1/sqrt(64)

  for (int mc = 0; mc < NN; mc += 32) {
    v16bf kb0a = load_frag(Kh + (size_t)mc * DH, DH, 0);
    v16bf kb0b = load_frag(Kh + (size_t)mc * DH, DH, 32);
    v16bf kb1a = load_frag(Kh + (size_t)(mc + 16) * DH, DH, 0);
    v16bf kb1b = load_frag(Kh + (size_t)(mc + 16) * DH, DH, 32);
    v8f s0 = {}; s0 = wmma_bf16(qa0, kb0a, s0); s0 = wmma_bf16(qa1, kb0b, s0);
    v8f s1 = {}; s1 = wmma_bf16(qa0, kb1a, s1); s1 = wmma_bf16(qa1, kb1b, s1);

    int c0 = mc + idx, c1 = mc + 16 + idx;
    int rc0 = (c0 >= ra) + (c0 >= rb);
    int rc1 = (c1 >= ra) + (c1 >= rb);

    float p0[8], p1v[8];
    for (int v = 0; v < 8; v++) {
      int r = r0 + halfb * 8 + v;
      size_t base = ((size_t)b * NN + r) * NN;
      float sv0 = s0[v] * scale - g[regr[v] * 4 + rc0] * Dd[base + c0] + NMd[base + c0];
      float sv1 = s1[v] * scale - g[regr[v] * 4 + rc1] * Dd[base + c1] + NMd[base + c1];
      float cm = fmaxf(sv0, sv1);
      cm = fmaxf(cm, __shfl_xor(cm, 1));
      cm = fmaxf(cm, __shfl_xor(cm, 2));
      cm = fmaxf(cm, __shfl_xor(cm, 4));
      cm = fmaxf(cm, __shfl_xor(cm, 8));
      float nM = fmaxf(Mx[v], cm);
      float e0 = __expf(sv0 - nM), e1 = __expf(sv1 - nM);
      float rs = e0 + e1;
      rs += __shfl_xor(rs, 1);
      rs += __shfl_xor(rs, 2);
      rs += __shfl_xor(rs, 4);
      rs += __shfl_xor(rs, 8);
      float alpha = __expf(Mx[v] - nM);
      L[v] = L[v] * alpha + rs;
      Mx[v] = nM;
      for (int j = 0; j < 4; j++) acc[j][v] *= alpha;
      p0[v]  = e0 * MSK[base + c0];   // mask applied after softmax numerator
      p1v[v] = e1 * MSK[base + c1];
    }

    // P tile (16x32) -> LDS -> WMMA A fragment (same-wave LDS ops are in order)
    for (int v = 0; v < 8; v++) {
      int m = halfb * 8 + v;
      plds[m * 32 + idx]      = f32_to_bf16(p0[v]);
      plds[m * 32 + 16 + idx] = f32_to_bf16(p1v[v]);
    }
    __syncthreads();
    v16bf pa = load_frag((const __bf16*)plds, 32, 0);
    for (int j = 0; j < 4; j++) {
      v16bf vb = load_frag(Vh + (size_t)(j * 16) * NN, NN, mc);
      acc[j] = wmma_bf16(pa, vb, acc[j]);
    }
    __syncthreads();
  }

  for (int v = 0; v < 8; v++) {
    float inv = 1.0f / L[v];
    int m = r0 + halfb * 8 + v;
    for (int j = 0; j < 4; j++) {
      float x = acc[j][v] * inv;
      x = (x > 0.f) ? x : 0.01f * x;  // leaky relu before O-projection
      attn_out[((size_t)(b * NN + m)) * DM + h * DH + j * 16 + idx] = f32_to_bf16(x);
    }
  }
}

// ---------------- O projection + residual (f32 out) ----------------
__global__ __launch_bounds__(32) void gemm_o_kernel(const __bf16* __restrict__ A,
                                                    const __bf16* __restrict__ W,
                                                    const float* __restrict__ Zin,
                                                    float* __restrict__ Zres) {
  int m0 = blockIdx.y * 16, n0 = blockIdx.x * 64;
  v8f acc[4] = {};
  for (int k = 0; k < DM; k += 32) {
    v16bf a = load_frag(A + (size_t)m0 * DM, DM, k);
    for (int j = 0; j < 4; j++)
      acc[j] = wmma_bf16(a, load_frag(W + (size_t)(n0 + j * 16) * DM, DM, k), acc[j]);
  }
  int lane = threadIdx.x & 31, halfb = lane >> 4, idx = lane & 15;
  for (int j = 0; j < 4; j++)
    for (int v = 0; v < 8; v++) {
      int m = m0 + halfb * 8 + v, col = n0 + j * 16 + idx;
      size_t off = (size_t)m * DM + col;
      Zres[off] = acc[j][v] + Zin[off];
    }
}

// ---------------- MLP up: relu(x@p1^T + b1) -> bf16 ----------------
__global__ __launch_bounds__(32) void gemm_p1_kernel(const __bf16* __restrict__ A,
                                                     const __bf16* __restrict__ W,
                                                     const float* __restrict__ bias,
                                                     __bf16* __restrict__ H1) {
  int m0 = blockIdx.y * 16, n0 = blockIdx.x * 64;
  v8f acc[4] = {};
  for (int k = 0; k < DM; k += 32) {
    v16bf a = load_frag(A + (size_t)m0 * DM, DM, k);
    for (int j = 0; j < 4; j++)
      acc[j] = wmma_bf16(a, load_frag(W + (size_t)(n0 + j * 16) * DM, DM, k), acc[j]);
  }
  int lane = threadIdx.x & 31, halfb = lane >> 4, idx = lane & 15;
  for (int j = 0; j < 4; j++)
    for (int v = 0; v < 8; v++) {
      int m = m0 + halfb * 8 + v, col = n0 + j * 16 + idx;
      float x = acc[j][v] + bias[col];
      x = (x > 0.f) ? x : 0.f;
      H1[(size_t)m * DI + col] = f32_to_bf16(x);
    }
}

// ---------------- MLP down: h1@p2^T + b2 + residual -> f32 out ----------------
__global__ __launch_bounds__(32) void gemm_p2_kernel(const __bf16* __restrict__ A,
                                                     const __bf16* __restrict__ W,
                                                     const float* __restrict__ bias,
                                                     const float* __restrict__ Zres,
                                                     float* __restrict__ out) {
  int m0 = blockIdx.y * 16, n0 = blockIdx.x * 64;
  v8f acc[4] = {};
  for (int k = 0; k < DI; k += 32) {
    v16bf a = load_frag(A + (size_t)m0 * DI, DI, k);
    for (int j = 0; j < 4; j++)
      acc[j] = wmma_bf16(a, load_frag(W + (size_t)(n0 + j * 16) * DI, DI, k), acc[j]);
  }
  int lane = threadIdx.x & 31, halfb = lane >> 4, idx = lane & 15;
  for (int j = 0; j < 4; j++)
    for (int v = 0; v < 8; v++) {
      int m = m0 + halfb * 8 + v, col = n0 + j * 16 + idx;
      size_t off = (size_t)m * DM + col;
      out[off] = acc[j][v] + bias[col] + Zres[off];
    }
}

// ---------------- host-side launch ----------------
extern "C" void kernel_launch(void* const* d_in, const int* in_sizes, int n_in,
                              void* d_out, int out_size, void* d_ws, size_t ws_size,
                              hipStream_t stream) {
  const float* Z     = (const float*)d_in[0];
  const float* Dd    = (const float*)d_in[1];
  const float* NMd   = (const float*)d_in[2];
  const float* MSK   = (const float*)d_in[3];
  const float* gamma = (const float*)d_in[4];
  const float* qkv_w = (const float*)d_in[5];
  const float* qkv_b = (const float*)d_in[6];
  const float* o_w   = (const float*)d_in[7];
  const float* ln1_w = (const float*)d_in[8];
  const float* ln1_b = (const float*)d_in[9];
  const float* ln2_w = (const float*)d_in[10];
  const float* ln2_b = (const float*)d_in[11];
  const float* p1_w  = (const float*)d_in[12];
  const float* p1_b  = (const float*)d_in[13];
  const float* p2_w  = (const float*)d_in[14];
  const float* p2_b  = (const float*)d_in[15];
  const int*   ra    = (const int*)d_in[16];
  const int*   rb    = (const int*)d_in[17];
  float* out = (float*)d_out;

  char* ws = (char*)d_ws;
  size_t off = 0;
  auto alloc = [&](size_t bytes) -> char* {
    char* p = ws + off;
    off += (bytes + 255) & ~(size_t)255;
    return p;
  };
  __bf16* qkvw_bf = (__bf16*)alloc((size_t)3 * DM * DM * 2);
  __bf16* ow_bf   = (__bf16*)alloc((size_t)DM * DM * 2);
  __bf16* p1w_bf  = (__bf16*)alloc((size_t)DI * DM * 2);
  __bf16* p2w_bf  = (__bf16*)alloc((size_t)DM * DI * 2);
  __bf16* Zn      = (__bf16*)alloc((size_t)ROWS * DM * 2);
  __bf16* Qb      = (__bf16*)alloc((size_t)BN * NH * NN * DH * 2);
  __bf16* Kb      = (__bf16*)alloc((size_t)BN * NH * NN * DH * 2);
  __bf16* Vt      = (__bf16*)alloc((size_t)BN * NH * NN * DH * 2);
  __bf16* attn_bf = (__bf16*)alloc((size_t)ROWS * DM * 2);
  float*  Zres    = (float*)alloc((size_t)ROWS * DM * 4);
  __bf16* Zn2     = (__bf16*)alloc((size_t)ROWS * DM * 2);
  __bf16* H1      = (__bf16*)alloc((size_t)ROWS * DI * 2);

  // 1) weight conversions
  {
    int n = 3 * DM * DM;
    cvt_bf16_kernel<<<(n + 255) / 256, 256, 0, stream>>>(qkv_w, qkvw_bf, n);
    n = DM * DM;
    cvt_bf16_kernel<<<(n + 255) / 256, 256, 0, stream>>>(o_w, ow_bf, n);
    n = DI * DM;
    cvt_bf16_kernel<<<(n + 255) / 256, 256, 0, stream>>>(p1_w, p1w_bf, n);
    n = DM * DI;
    cvt_bf16_kernel<<<(n + 255) / 256, 256, 0, stream>>>(p2_w, p2w_bf, n);
  }
  // 2) LN1
  ln_kernel<<<ROWS, 256, 0, stream>>>(Z, ln1_w, ln1_b, Zn);
  // 3) QKV projection
  gemm_qkv_kernel<<<dim3(3 * DM / 64, ROWS / 16), 32, 0, stream>>>(Zn, qkvw_bf, qkv_b, Qb, Kb, Vt);
  // 4) attention
  attn_kernel<<<dim3(NN / 16, NH, BN), 32, 0, stream>>>(Qb, Kb, Vt, Dd, NMd, MSK, gamma, ra, rb, attn_bf);
  // 5) O projection + residual
  gemm_o_kernel<<<dim3(DM / 64, ROWS / 16), 32, 0, stream>>>(attn_bf, ow_bf, Z, Zres);
  // 6) LN2
  ln_kernel<<<ROWS, 256, 0, stream>>>(Zres, ln2_w, ln2_b, Zn2);
  // 7) MLP up
  gemm_p1_kernel<<<dim3(DI / 64, ROWS / 16), 32, 0, stream>>>(Zn2, p1w_bf, p1_b, H1);
  // 8) MLP down + residual -> out
  gemm_p2_kernel<<<dim3(DM / 64, ROWS / 16), 32, 0, stream>>>(H1, p2w_bf, p2_b, Zres, out);
}
